// ChannelAttentionModule_9320079032515
// MI455X (gfx1250) — matrix operand order repaired and checked by reading, not compiled, not run on hardware
//
#include <hip/hip_runtime.h>

// ---------------------------------------------------------------------------
// Channel attention (DANet-style) for x[16,512,64,64] f32 on MI455X (gfx1250).
//   S = F F^T  (per batch, 512x512x4096)   -> bf16 WMMA, f32 accumulate
//   G = softmax(S_max - S) rowwise         -> = exp(S_min - S)/sum(...)
//   out = beta * (G F) + x                 -> bf16 WMMA, fused epilogue
// v_wmma_f32_16x16x32_bf16 (K=32) instead of fp32 16x16x4 (8x denser math,
// f32 accumulation preserved). Staging uses GLOBAL_LOAD_ASYNC_TO_LDS_B128
// (ASYNCcnt) with double-buffered LDS so copies for step k+1 overlap the
// WMMAs of step k (1 barrier/step).
// ---------------------------------------------------------------------------

typedef __bf16 bf16_t;
typedef __attribute__((ext_vector_type(16))) __bf16 v16bf;
typedef __attribute__((ext_vector_type(8)))  float  v8f;
typedef __attribute__((ext_vector_type(4)))  int    v4i;

#define BATCH 16
#define CCH   512
#define NPIX  4096
#define KT    32          // K elements per mainloop step (matches WMMA K)
#define BM    128         // workgroup tile M
#define BN    128         // workgroup tile N
#define SROW  40          // LDS row stride in halves: 32 + 8 pad (80B, 16B aligned)

#if __has_builtin(__builtin_amdgcn_global_load_async_to_lds_b128) && \
    __has_builtin(__builtin_amdgcn_s_wait_asynccnt)
#define HAVE_ASYNC_LDS 1
// builtin prototype (from hipcc diagnostic): arg0 = v4i in global AS (AS1),
// arg1 = v4i in LDS AS (AS3), then imm offset, imm cpol.
typedef __attribute__((address_space(1))) v4i* gv4ip;
typedef __attribute__((address_space(3))) v4i* lv4ip;
#else
#define HAVE_ASYNC_LDS 0
#endif

union FragU  { struct { uint4 lo, hi; } q; v16bf v; };
union Pack8  { uint4 u; bf16_t h[8]; };

__device__ __forceinline__ v8f zero8() {
  v8f z;
  #pragma unroll
  for (int e = 0; e < 8; ++e) z[e] = 0.0f;
  return z;
}

// 16B global -> LDS copy: async (no VGPR roundtrip) when available.
__device__ __forceinline__ void copy16_to_lds(bf16_t* lds_dst, const bf16_t* gsrc) {
#if HAVE_ASYNC_LDS
  __builtin_amdgcn_global_load_async_to_lds_b128((gv4ip)gsrc, (lv4ip)lds_dst, 0, 0);
#else
  *(uint4*)lds_dst = *(const uint4*)gsrc;
#endif
}

__device__ __forceinline__ void wait_async_copies() {
#if HAVE_ASYNC_LDS
  __builtin_amdgcn_s_wait_asynccnt(0);
#endif
}

// ---------------------------------------------------------------- convert ---
__global__ __launch_bounds__(256)
void cam_cvt_bf16(const float* __restrict__ x, bf16_t* __restrict__ xb) {
  const long long i = (long long)blockIdx.x * blockDim.x + threadIdx.x; // one float4
  const float4 v = ((const float4*)x)[i];
  union { bf16_t h[4]; uint2 u; } o;
  o.h[0] = (bf16_t)v.x; o.h[1] = (bf16_t)v.y;
  o.h[2] = (bf16_t)v.z; o.h[3] = (bf16_t)v.w;
  ((uint2*)xb)[i] = o.u;
}

// ------------------------------------------------------------- Gram GEMM ---
// S[b] = F F^T.  A panel = rows bm0.., B^T panel = rows bn0.. of F (both
// row-major in global). Double-buffered async LDS staging.
__global__ __launch_bounds__(256)
void cam_gram_wmma(const bf16_t* __restrict__ xb, float* __restrict__ S) {
  __shared__ bf16_t Asm[2][BM * SROW];
  __shared__ bf16_t Bsm[2][BN * SROW];

  const int b   = blockIdx.z;
  const int bm0 = blockIdx.y * BM;
  const int bn0 = blockIdx.x * BN;
  const int t    = threadIdx.x;
  const int wave = t >> 5;
  const int lane = t & 31;
  const int mw = (wave & 3) * 32;   // wave sub-tile: 32 rows x 64 cols
  const int nw = (wave >> 2) * 64;
  const int lm   = lane & 15;
  const int ak   = (lane < 16) ? 0 : 16;   // A frag K base (halves)
  const int bk   = (lane < 16) ? 0 : 16;   // B frag K base (halves)
  const int moff = (lane < 16) ? 0 : 8;    // C/D row offset

  const bf16_t* Abase = xb + ((size_t)b * CCH + bm0) * NPIX;
  const bf16_t* Bbase = xb + ((size_t)b * CCH + bn0) * NPIX;

  // staging map: thread t copies 2x16B per panel (128 rows x 32 halves)
  const int srow = t >> 1;
  const int sch  = (t & 1) * 2;   // chunks sch, sch+1 (each 8 halves)
  const int soff = srow * SROW + sch * 8;

  v8f acc[2][4];
  #pragma unroll
  for (int i = 0; i < 2; ++i)
    #pragma unroll
    for (int j = 0; j < 4; ++j) acc[i][j] = zero8();

  auto stage = [&](int sel, int k0) {
    const bf16_t* ag = Abase + (size_t)srow * NPIX + k0 + sch * 8;
    const bf16_t* bg = Bbase + (size_t)srow * NPIX + k0 + sch * 8;
    copy16_to_lds(&Asm[sel][soff],     ag);
    copy16_to_lds(&Asm[sel][soff + 8], ag + 8);
    copy16_to_lds(&Bsm[sel][soff],     bg);
    copy16_to_lds(&Bsm[sel][soff + 8], bg + 8);
  };

  int sel = 0;
  stage(sel, 0);
  wait_async_copies();
  __syncthreads();

  for (int k0 = 0; k0 < NPIX; k0 += KT) {
    const int nxt = sel ^ 1;
    if (k0 + KT < NPIX) stage(nxt, k0 + KT);   // overlap copy with compute

    FragU af[2], bf[4];
    #pragma unroll
    for (int i = 0; i < 2; ++i) {
      const bf16_t* p = &Asm[sel][(mw + i * 16 + lm) * SROW + ak];
      af[i].q.lo = *(const uint4*)p;
      af[i].q.hi = *(const uint4*)(p + 8);
    }
    #pragma unroll
    for (int j = 0; j < 4; ++j) {
      const bf16_t* p = &Bsm[sel][(nw + j * 16 + lm) * SROW + bk];
      bf[j].q.lo = *(const uint4*)p;
      bf[j].q.hi = *(const uint4*)(p + 8);
    }
    #pragma unroll
    for (int i = 0; i < 2; ++i)
      #pragma unroll
      for (int j = 0; j < 4; ++j)
        acc[i][j] = __builtin_amdgcn_wmma_f32_16x16x32_bf16(
            false, af[i].v, false, bf[j].v, (short)0, acc[i][j], false, false);

    wait_async_copies();
    __syncthreads();
    sel = nxt;
  }

  float* Sb = S + (size_t)b * CCH * CCH;
  #pragma unroll
  for (int i = 0; i < 2; ++i)
    #pragma unroll
    for (int j = 0; j < 4; ++j)
      #pragma unroll
      for (int v = 0; v < 8; ++v) {
        const int r = bm0 + mw + i * 16 + v + moff;
        const int c = bn0 + nw + j * 16 + lm;
        Sb[(size_t)r * CCH + c] = acc[i][j][v];
      }
}

// ------------------------------------------------------------ row softmax ---
// G[r,:] = softmax(S_max - S)[r,:] == exp(S_min[r]-S)/sum(exp(S_min[r]-S))
__global__ __launch_bounds__(256)
void cam_rowsoftmax(const float* __restrict__ S, bf16_t* __restrict__ G) {
  __shared__ float red[256];
  const size_t base = (size_t)blockIdx.x * CCH;
  const int t = threadIdx.x;
  const float s0 = S[base + t];
  const float s1 = S[base + t + 256];
  red[t] = fminf(s0, s1);
  __syncthreads();
  #pragma unroll
  for (int s = 128; s > 0; s >>= 1) {
    if (t < s) red[t] = fminf(red[t], red[t + s]);
    __syncthreads();
  }
  const float rmin = red[0];
  __syncthreads();
  const float e0 = __expf(rmin - s0);
  const float e1 = __expf(rmin - s1);
  red[t] = e0 + e1;
  __syncthreads();
  #pragma unroll
  for (int s = 128; s > 0; s >>= 1) {
    if (t < s) red[t] += red[t + s];
    __syncthreads();
  }
  const float inv = 1.0f / red[0];
  G[base + t]       = (bf16_t)(e0 * inv);
  G[base + t + 256] = (bf16_t)(e1 * inv);
}

// ------------------------------------------- attention apply + epilogue ----
// out = beta * (G F) + x.  A = G (row-major, ld=C, async-staged).  B = F
// ([K=C, N=NPIX]); staged transposed into LDS (VGPR path — async copies
// cannot transpose) so fragment reads stay contiguous b128s. Both overlap
// the WMMAs via double buffering.
__global__ __launch_bounds__(256)
void cam_apply_wmma(const bf16_t* __restrict__ G, const bf16_t* __restrict__ xb,
                    const float* __restrict__ x, const float* __restrict__ betap,
                    float* __restrict__ out) {
  __shared__ bf16_t Asm[2][BM * SROW];
  __shared__ bf16_t Bsm[2][BN * SROW];

  const int b   = blockIdx.z;
  const int bm0 = blockIdx.y * BM;   // channel rows
  const int bn0 = blockIdx.x * BN;   // pixel cols
  const int t    = threadIdx.x;
  const int wave = t >> 5;
  const int lane = t & 31;
  const int mw = (wave & 3) * 32;
  const int nw = (wave >> 2) * 64;
  const int lm   = lane & 15;
  const int ak   = (lane < 16) ? 0 : 16;
  const int bk   = (lane < 16) ? 0 : 16;
  const int moff = (lane < 16) ? 0 : 8;

  const bf16_t* Abase = G  + ((size_t)b * CCH + bm0) * CCH;
  const bf16_t* Bbase = xb + (size_t)b * CCH * NPIX;

  const int srow = t >> 1;
  const int sch  = (t & 1) * 2;
  const int soff = srow * SROW + sch * 8;
  const int kl0  = t >> 4;          // B staging: k row 0..15 (+16 on pass 2)
  const int nb   = (t & 15) * 8;    // B staging: 8 consecutive n

  v8f acc[2][4];
  #pragma unroll
  for (int i = 0; i < 2; ++i)
    #pragma unroll
    for (int j = 0; j < 4; ++j) acc[i][j] = zero8();

  auto stage = [&](int sel, int k0) {
    const bf16_t* ag = Abase + (size_t)srow * CCH + k0 + sch * 8;
    copy16_to_lds(&Asm[sel][soff],     ag);
    copy16_to_lds(&Asm[sel][soff + 8], ag + 8);
    #pragma unroll
    for (int p = 0; p < 2; ++p) {   // transpose-stage B: coalesced global read
      const int kl = kl0 + p * 16;
      Pack8 ld;
      ld.u = *(const uint4*)(Bbase + (size_t)(k0 + kl) * NPIX + bn0 + nb);
      #pragma unroll
      for (int j = 0; j < 8; ++j) Bsm[sel][(nb + j) * SROW + kl] = ld.h[j];
    }
  };

  int sel = 0;
  stage(sel, 0);
  wait_async_copies();
  __syncthreads();

  for (int k0 = 0; k0 < CCH; k0 += KT) {
    const int nxt = sel ^ 1;
    if (k0 + KT < CCH) stage(nxt, k0 + KT);

    FragU af[2], bf[4];
    #pragma unroll
    for (int i = 0; i < 2; ++i) {
      const bf16_t* p = &Asm[sel][(mw + i * 16 + lm) * SROW + ak];
      af[i].q.lo = *(const uint4*)p;
      af[i].q.hi = *(const uint4*)(p + 8);
    }
    #pragma unroll
    for (int j = 0; j < 4; ++j) {
      const bf16_t* p = &Bsm[sel][(nw + j * 16 + lm) * SROW + bk];
      bf[j].q.lo = *(const uint4*)p;
      bf[j].q.hi = *(const uint4*)(p + 8);
    }
    #pragma unroll
    for (int i = 0; i < 2; ++i)
      #pragma unroll
      for (int j = 0; j < 4; ++j)
        acc[i][j] = __builtin_amdgcn_wmma_f32_16x16x32_bf16(
            false, af[i].v, false, bf[j].v, (short)0, acc[i][j], false, false);

    wait_async_copies();
    __syncthreads();
    sel = nxt;
  }

  const float beta = betap[0];
  #pragma unroll
  for (int i = 0; i < 2; ++i)
    #pragma unroll
    for (int j = 0; j < 4; ++j)
      #pragma unroll
      for (int v = 0; v < 8; ++v) {
        const int r = bm0 + mw + i * 16 + v + moff;
        const int c = bn0 + nw + j * 16 + lm;
        const size_t idx = ((size_t)b * CCH + r) * NPIX + c;
        out[idx] = beta * acc[i][j][v] + x[idx];
      }
}

// ---------------------------------------------------------------- launch ---
extern "C" void kernel_launch(void* const* d_in, const int* in_sizes, int n_in,
                              void* d_out, int out_size, void* d_ws, size_t ws_size,
                              hipStream_t stream) {
  const float* x    = (const float*)d_in[0];
  const float* beta = (const float*)d_in[1];
  float* out = (float*)d_out;

  // workspace layout: xb (bf16, 64MB) | S (f32, 16MB) | G (bf16, 8MB)
  bf16_t* xb = (bf16_t*)d_ws;
  float*  S  = (float*)((char*)d_ws + (size_t)BATCH * CCH * NPIX * sizeof(bf16_t));
  bf16_t* G  = (bf16_t*)((char*)S + (size_t)BATCH * CCH * CCH * sizeof(float));

  const long long n4 = (long long)BATCH * CCH * NPIX / 4;  // float4 count
  cam_cvt_bf16<<<(unsigned)(n4 / 256), 256, 0, stream>>>(x, xb);

  cam_gram_wmma<<<dim3(CCH / BN, CCH / BM, BATCH), 256, 0, stream>>>(xb, S);

  cam_rowsoftmax<<<BATCH * CCH, 256, 0, stream>>>(S, G);

  cam_apply_wmma<<<dim3(NPIX / BN, CCH / BM, BATCH), 256, 0, stream>>>(
      G, xb, x, beta, out);
}
